// VQPPF_29429115912771
// MI455X (gfx1250) — matrix operand order repaired and checked by reading, not compiled, not run on hardware
//
#include <hip/hip_runtime.h>

// ---------------------------------------------------------------------------
// VQPPF forward on gfx1250 (MI455X). All convs / conv-transposes / VQ GEMMs
// run as implicit GEMM through v_wmma_f32_16x16x32_f16 (fp16 in, fp32 acc),
// with LDS-staged fragments pre-permuted into contiguous-k per-lane order (k = 16*(lane/16) + e),
// 2 WMMAs per barrier round, and register prefetch of the next K slice.
// All spatial geometry is power-of-two -> pure shift/mask addressing.
// ---------------------------------------------------------------------------

typedef __attribute__((ext_vector_type(16))) _Float16 v16h;
typedef __attribute__((ext_vector_type(8)))  _Float16 v8h;
typedef __attribute__((ext_vector_type(4)))  _Float16 v4h;
typedef __attribute__((ext_vector_type(8)))  float    v8f;

#define CF_RELU_IN  2   // apply relu when reading input (pre-activation)
#define CF_RELU_OUT 4   // apply relu to output

// Implicit-GEMM conv / conv-transpose.
//   D[M=Cout x N=B*OH*OW] = W[M x K] * im2col(in)[K x N],  K = Cin*KH*KW
// Block = 256 threads = 8 waves; block tile 32(M) x 64(N); wave (mi,ni) owns
// one 16x16 D tile. Per 64-wide K round, two 32-K fragment sets (A 32x32,
// B 32x64 each) are cooperatively gathered (fp32->f16) into LDS in contiguous-k (16h+e)
// fragment order; each lane then reads each fragment as one v16h and issues
// two WMMAs, while the next round's gathers are prefetched into registers.
// IH==IW==1<<inShift, OH==OW==1<<hwShift (all layers are square pow2).
template <int KH, int KW, int STRIDE, bool TRANS>
__global__ __launch_bounds__(256) void conv_wmma_kernel(
    const float* __restrict__ in, const float* __restrict__ w,
    const float* __restrict__ bias, const float* __restrict__ res,
    float* __restrict__ out,
    int B, int Cin, int inShift, int Cout, int hwShift, int pad, int flags)
{
    constexpr int KHW = KH * KW;
    const int IH = 1 << inShift;
    const int OH = 1 << hwShift;
    const int hwMask = OH - 1;
    __shared__ __align__(16) _Float16 aLds[2][2][32][16];   // [chunk][mi][lane][e] 4KB
    __shared__ __align__(16) _Float16 bLds[2][4][32][16];   // [chunk][ni][lane][e] 8KB

    const int t    = threadIdx.x;
    const int lane = t & 31;
    const int wave = t >> 5;
    const int miW  = wave >> 2;          // 0..1  (M subtile)
    const int niW  = wave & 3;           // 0..3  (N subtile)
    const int coB  = blockIdx.y * 32;
    const int posB = blockIdx.x * 64;
    const int K    = Cin * KHW;
    const int Npos = B << (2 * hwShift);
    const bool rin = (flags & CF_RELU_IN) != 0;

    // ---- A staging geometry: this thread owns aLds[c][sa_mi][sa_l][sa_e0..+3]
    const int sa_e0 = (t & 3) * 4;
    const int sa_l  = (t >> 2) & 31;
    const int sa_mi = t >> 7;
    const int sa_co = coB + sa_mi * 16 + (sa_l & 15);
    const int sa_kb = (sa_l >> 4) * 16;
    const bool sa_cov = sa_co < Cout;

    // ---- B staging geometry: this thread owns bLds[c][sb_ni][sb_l][sb_e0..+7]
    const int sb_e0 = (t & 1) * 8;
    const int sb_l  = (t >> 1) & 31;
    const int sb_ni = t >> 6;
    const int sb_kb = (sb_l >> 4) * 16;
    const int sb_pos = posB + sb_ni * 16 + (sb_l & 15);
    const bool sb_v = sb_pos < Npos;
    const int sb_ox = sb_pos & hwMask;
    const int sb_oy = (sb_pos >> hwShift) & hwMask;
    const int sb_b  = sb_pos >> (2 * hwShift);
    const int sb_inB = sb_b * Cin;

    // ---- consumer epilogue geometry
    const int chi  = lane >> 4;
    const int cpos = posB + niW * 16 + (lane & 15);
    const bool cposv = cpos < Npos;
    const int cox = cpos & hwMask;
    const int coy = (cpos >> hwShift) & hwMask;
    const int cb  = cpos >> (2 * hwShift);

    float aReg[2][4], bReg[2][8];

    auto fetchA = [&](int k0, float* r) {
        #pragma unroll
        for (int i = 0; i < 4; ++i) {
            int e = sa_e0 + i;
            int k = k0 + sa_kb + e;
            float v = 0.f;
            if (sa_cov && k < K) {
                int ci = k / KHW; int rr = k - ci * KHW;    // const divisor
                int ky = rr / KW; int kx = rr - ky * KW;
                v = TRANS ? w[((ci * Cout + sa_co) * KH + ky) * KW + kx]
                          : w[((sa_co * Cin + ci) * KH + ky) * KW + kx];
            }
            r[i] = v;
        }
    };
    auto fetchB = [&](int k0, float* r) {
        #pragma unroll
        for (int i = 0; i < 8; ++i) {
            int k = k0 + sb_kb + sb_e0 + i;
            float v = 0.f;
            if (sb_v && k < K) {
                int ci = k / KHW; int rr = k - ci * KHW;
                int ky = rr / KW; int kx = rr - ky * KW;
                int iy, ix; bool ok;
                if (!TRANS) {
                    iy = sb_oy * STRIDE - pad + ky;
                    ix = sb_ox * STRIDE - pad + kx;
                    ok = ((unsigned)iy < (unsigned)IH) & ((unsigned)ix < (unsigned)IH);
                } else {
                    int ty = sb_oy + pad - ky;
                    int tx = sb_ox + pad - kx;
                    iy = ty / STRIDE; ix = tx / STRIDE;
                    ok = (ty >= 0) && (tx >= 0) && (iy * STRIDE == ty) &&
                         (ix * STRIDE == tx) && (iy < IH) && (ix < IH);
                }
                if (ok) {
                    v = in[((sb_inB + ci) << (2 * inShift)) + (iy << inShift) + ix];
                    if (rin) v = fmaxf(v, 0.f);
                }
            }
            r[i] = v;
        }
    };

    v8f acc = {};
    fetchA(0,  aReg[0]); fetchA(32, aReg[1]);
    fetchB(0,  bReg[0]); fetchB(32, bReg[1]);
    for (int k0 = 0; k0 < K; k0 += 64) {
        // commit both staged 32-K chunks to LDS (vector DS stores)
        #pragma unroll
        for (int c = 0; c < 2; ++c) {
            v4h av; v8h bv;
            #pragma unroll
            for (int i = 0; i < 4; ++i) av[i] = (_Float16)aReg[c][i];
            #pragma unroll
            for (int i = 0; i < 8; ++i) bv[i] = (_Float16)bReg[c][i];
            *(v4h*)(&aLds[c][sa_mi][sa_l][sa_e0]) = av;   // ds_store_b64
            *(v8h*)(&bLds[c][sb_ni][sb_l][sb_e0]) = bv;   // ds_store_b128
        }
        __syncthreads();
        int kn = k0 + 64;
        if (kn < K) {                                     // prefetch under WMMA
            fetchA(kn,      aReg[0]); fetchA(kn + 32, aReg[1]);
            fetchB(kn,      bReg[0]); fetchB(kn + 32, bReg[1]);
        }
        #pragma unroll
        for (int c = 0; c < 2; ++c) {
            v16h af = *(const v16h*)(&aLds[c][miW][lane][0]);
            v16h bf = *(const v16h*)(&bLds[c][niW][lane][0]);
            acc = __builtin_amdgcn_wmma_f32_16x16x32_f16(
                      false, af, false, bf, (short)0, acc, false, false);
        }
        __syncthreads();
    }

    // ---- epilogue: C/D layout (lane = col, vgpr v = row) -> NCHW
    #pragma unroll
    for (int v = 0; v < 8; ++v) {
        int co = coB + miW * 16 + v + chi * 8;
        if (co < Cout && cposv) {
            float val = acc[v];
            if (bias) val += bias[co];
            int oidx = ((cb * Cout + co) << (2 * hwShift)) + (coy << hwShift) + cox;
            if (res)  val += res[oidx];
            if (flags & CF_RELU_OUT) val = fmaxf(val, 0.f);
            out[oidx] = val;
        }
    }
}

// ---------------------------------------------------------------------------
// Support kernels (hard-coded pow2 geometry: spatial=4096, edim=64, codes=512)
// ---------------------------------------------------------------------------

__global__ void zero_kernel(float* acc, unsigned* counts) {
    int t = threadIdx.x;
    if (t < 64) acc[t] = 0.f;
    for (int i = t; i < 2048; i += 256) counts[i] = 0u;
}

__global__ void esq_kernel(const float* __restrict__ emb, float* __restrict__ esq) {
    int e = blockIdx.x * 256 + threadIdx.x;   // 512 codes
    if (e < 512) {
        float s = 0.f;
        #pragma unroll 8
        for (int c = 0; c < 64; ++c) { float v = emb[e * 64 + c]; s += v * v; }
        esq[e] = s;
    }
}

// argmin_e (||e||^2 - 2<z,e>) per position + histogram. scores: (4,512,64,64)
__global__ void vq_argmin_kernel(const float* __restrict__ scores,
                                 const float* __restrict__ esq,
                                 int* __restrict__ idx, unsigned* __restrict__ counts) {
    int p = blockIdx.x * 256 + threadIdx.x;   // 16384 positions
    int b = p >> 12, yx = p & 4095;
    const float* s = scores + ((size_t)b << 21) + yx;
    float best = 3.4e38f; int bi = 0;
    #pragma unroll 4
    for (int e = 0; e < 512; ++e) {
        float d = esq[e] - 2.0f * s[(size_t)e << 12];
        if (d < best) { best = d; bi = e; }   // first-min tie-break == argmin
    }
    idx[p] = bi;
    atomicAdd(&counts[bi], 1u);
}

// z_q gather into NCHW + accumulate sum((z_q - z)^2). lat/zq: (4,64,64,64)
__global__ void vq_gather_kernel(const float* __restrict__ lat,
                                 const float* __restrict__ emb,
                                 const int* __restrict__ idx,
                                 float* __restrict__ zq, float* __restrict__ acc) {
    int t = blockIdx.x * 256 + threadIdx.x;   // 1048576 elements
    int p = t >> 6, c = t & 63;
    int b = p >> 12, yx = p & 4095;
    int li = ((b << 6) + c) * 4096 + yx;
    float zv = lat[li];
    float qv = emb[(idx[p] << 6) + c];
    zq[li] = qv;
    float d = qv - zv;
    float mine = d * d;
    __shared__ float red[256];
    red[threadIdx.x] = mine; __syncthreads();
    for (int s = 128; s > 0; s >>= 1) {
        if (threadIdx.x < s) red[threadIdx.x] += red[threadIdx.x + s];
        __syncthreads();
    }
    if (threadIdx.x == 0) atomicAdd(acc, red[0]);
}

__global__ void fusion_kernel(const float* __restrict__ zq,
                              const float* __restrict__ mu1,
                              const float* __restrict__ mu2,
                              float* __restrict__ out, int n) {
    int t = blockIdx.x * 256 + threadIdx.x;
    if (t >= n) return;
    out[t] = mu1[0] * zq[t] + mu2[0] * (zq[t + n] + zq[t + 2 * n] + zq[t + 3 * n]);
}

__global__ void mse_kernel(const float* __restrict__ a, const float* __restrict__ b,
                           float* __restrict__ acc, int n) {
    int t = blockIdx.x * 256 + threadIdx.x;
    float mine = 0.f;
    if (t < n) { float d = a[t] - b[t]; mine = d * d; }
    __shared__ float red[256];
    red[threadIdx.x] = mine; __syncthreads();
    for (int s = 128; s > 0; s >>= 1) {
        if (threadIdx.x < s) red[threadIdx.x] += red[threadIdx.x + s];
        __syncthreads();
    }
    if (threadIdx.x == 0) atomicAdd(acc, red[0]);
}

// total + perplexity. acc: [0]=vq sumsq, [1]=L_t, [2]=L_z, [3,4]=L_d sums
__global__ void finalize_kernel(const float* __restrict__ acc,
                                const unsigned* __restrict__ counts,
                                float* __restrict__ out) {
    __shared__ float red[256];
    __shared__ float perp_sum;
    if (threadIdx.x == 0) perp_sum = 0.f;
    __syncthreads();
    for (int i = 0; i < 4; ++i) {
        float h = 0.f;
        for (int e = threadIdx.x; e < 512; e += 256) {
            float p = (float)counts[i * 512 + e] * (1.0f / 16384.0f);
            h -= p * logf(p + 1e-10f);
        }
        red[threadIdx.x] = h; __syncthreads();
        for (int s = 128; s > 0; s >>= 1) {
            if (threadIdx.x < s) red[threadIdx.x] += red[threadIdx.x + s];
            __syncthreads();
        }
        if (threadIdx.x == 0) perp_sum += expf(red[0]);
        __syncthreads();
    }
    if (threadIdx.x == 0) {
        float total = acc[1] * (1.f / 196608.f)                  // L_t
                    + acc[2] * (1.f / 196608.f)                  // L_z
                    + (acc[3] + acc[4]) * (1.f / 786432.f) / 3.f // L_d / 3
                    + 1.25f * acc[0] * (1.f / 1048576.f);        // (1+beta)*vq
        out[0] = total;
        out[1] = perp_sum * 0.25f;
    }
}

// ---------------------------------------------------------------------------
// Host orchestration
// ---------------------------------------------------------------------------

extern "C" void kernel_launch(void* const* d_in, const int* in_sizes, int n_in,
                              void* d_out, int out_size, void* d_ws, size_t ws_size,
                              hipStream_t stream)
{
    const float* Zin  = (const float*)d_in[0];
    const float* X1   = (const float*)d_in[1];
    const float* X2   = (const float*)d_in[2];
    const float* X3   = (const float*)d_in[3];
    const float* ZTV  = (const float*)d_in[4];
    const float* EZ_C1W=(const float*)d_in[5],  *EZ_C1B=(const float*)d_in[6];
    const float* EZ_C2W=(const float*)d_in[7],  *EZ_C2B=(const float*)d_in[8];
    const float* EZ_C3W=(const float*)d_in[9],  *EZ_C3B=(const float*)d_in[10];
    const float* EZ_R1 =(const float*)d_in[11], *EZ_R2 =(const float*)d_in[12];
    const float* EX_C1W=(const float*)d_in[13], *EX_C1B=(const float*)d_in[14];
    const float* EX_C2W=(const float*)d_in[15], *EX_C2B=(const float*)d_in[16];
    const float* EX_C3W=(const float*)d_in[17], *EX_C3B=(const float*)d_in[18];
    const float* EX_R1 =(const float*)d_in[19], *EX_R2 =(const float*)d_in[20];
    const float* PQW  = (const float*)d_in[21], *PQB = (const float*)d_in[22];
    const float* EMB  = (const float*)d_in[23];
    const float* EMBZ = (const float*)d_in[24];
    const float* D_T1W=(const float*)d_in[25], *D_T1B=(const float*)d_in[26];
    const float* D_R1 =(const float*)d_in[27], *D_R2 =(const float*)d_in[28];
    const float* D_T2W=(const float*)d_in[29], *D_T2B=(const float*)d_in[30];
    const float* D_T3W=(const float*)d_in[31], *D_T3B=(const float*)d_in[32];
    const float* DZ_T1W=(const float*)d_in[33], *DZ_T1B=(const float*)d_in[34];
    const float* DZ_R1 =(const float*)d_in[35], *DZ_R2 =(const float*)d_in[36];
    const float* DZ_T2W=(const float*)d_in[37], *DZ_T2B=(const float*)d_in[38];
    const float* DZ_T3W=(const float*)d_in[39], *DZ_T3B=(const float*)d_in[40];
    const float* DX_T1W=(const float*)d_in[41], *DX_T1B=(const float*)d_in[42];
    const float* DX_R1 =(const float*)d_in[43], *DX_R2 =(const float*)d_in[44];
    const float* DX_T2W=(const float*)d_in[45], *DX_T2B=(const float*)d_in[46];
    const float* DX_T3W=(const float*)d_in[47], *DX_T3B=(const float*)d_in[48];
    const float* MU1  = (const float*)d_in[49];
    const float* MU2  = (const float*)d_in[50];

    // Workspace bump allocation (floats)
    float* ws   = (float*)d_ws;
    float* ACC  = ws;                          // 64 accumulators
    float* ESQZ = ws + 64;                     // 512
    float* ESQ  = ws + 576;                    // 512
    unsigned* COUNTS = (unsigned*)(ws + 1088); // 4*512
    int*   IDX  = (int*)(ws + 3136);           // 16384
    float* BUF0 = ws + 32768;                  // 16,777,216  (4,256,128,128)
    float* BUF1 = BUF0 + 16777216;             //  8,388,608  scores (4,512,64,64)
    float* BUF2 = BUF1 + 8388608;              //  4,194,304  (4,256,64,64)
    float* BUF3 = BUF2 + 4194304;              //  2,097,152  (4,128,64,64)
    float* BUF4 = BUF3 + 2097152;              //  2,097,152
    float* BUF5 = BUF4 + 2097152;              //  2,097,152
    float* LAT  = BUF5 + 2097152;              //  4 x 1,048,576 latents
    float* FUS  = LAT + 4194304;               //  1,048,576

    float* OUT = (float*)d_out;
    float* ZQ  = OUT + 2;                      // 4 x (4,64,64,64) contiguous

    auto lg2 = [](int v) { int s = 0; while ((1 << s) < v) ++s; return s; };

    auto conv = [&](const float* in, const float* w, const float* bias,
                    const float* res, float* o,
                    int B, int Cin, int IH, int Cout, int OH,
                    int KH, int s, int p, bool trans, int flags) {
        int npos = B * OH * OH;
        int inS = lg2(IH), hwS = lg2(OH);
        dim3 grid((unsigned)((npos + 63) / 64), (unsigned)((Cout + 31) / 32), 1);
        if (trans) {
            if (KH == 3)
                conv_wmma_kernel<3,3,1,true><<<grid, 256, 0, stream>>>(
                    in, w, bias, res, o, B, Cin, inS, Cout, hwS, p, flags);
            else
                conv_wmma_kernel<4,4,2,true><<<grid, 256, 0, stream>>>(
                    in, w, bias, res, o, B, Cin, inS, Cout, hwS, p, flags);
        } else {
            if (KH == 4)
                conv_wmma_kernel<4,4,2,false><<<grid, 256, 0, stream>>>(
                    in, w, bias, res, o, B, Cin, inS, Cout, hwS, p, flags);
            else if (KH == 3)
                conv_wmma_kernel<3,3,1,false><<<grid, 256, 0, stream>>>(
                    in, w, bias, res, o, B, Cin, inS, Cout, hwS, p, flags);
            else
                conv_wmma_kernel<1,1,1,false><<<grid, 256, 0, stream>>>(
                    in, w, bias, res, o, B, Cin, inS, Cout, hwS, p, flags);
        }
    };

    zero_kernel<<<1, 256, 0, stream>>>(ACC, COUNTS);
    esq_kernel<<<2, 256, 0, stream>>>(EMBZ, ESQZ);
    esq_kernel<<<2, 256, 0, stream>>>(EMB,  ESQ);

    // ---------------- Encoders ----------------
    auto encoder = [&](const float* xin, int IH,
                       const float* c1w, const float* c1b,
                       const float* c2w, const float* c2b, int k2, int s2,
                       const float* c3w, const float* c3b,
                       const float* r1, const float* r2, float* latout) {
        int H1 = IH / 2;
        float* o1 = (IH == 256) ? BUF0 : BUF2;
        conv(xin, c1w, c1b, nullptr, o1, 4, 3, IH, 256, H1, 4, 2, 1, false, CF_RELU_OUT);
        conv(o1, c2w, c2b, nullptr, BUF3, 4, 256, H1, 128, 64, k2, s2, 1, false, CF_RELU_OUT);
        conv(BUF3, c3w, c3b, nullptr, BUF4, 4, 128, 64, 128, 64, 3, 1, 1, false, 0);
        // res stack (shared r1/r2, 2 iterations); trailing relu folded into pre_q
        conv(BUF4, r1, nullptr, nullptr, BUF5, 4, 128, 64, 32, 64, 3, 1, 1, false, CF_RELU_IN);
        conv(BUF5, r2, nullptr, BUF4, BUF3, 4, 32, 64, 128, 64, 1, 1, 0, false, CF_RELU_IN);
        conv(BUF3, r1, nullptr, nullptr, BUF5, 4, 128, 64, 32, 64, 3, 1, 1, false, CF_RELU_IN);
        conv(BUF5, r2, nullptr, BUF3, BUF4, 4, 32, 64, 128, 64, 1, 1, 0, false, CF_RELU_IN);
        conv(BUF4, PQW, PQB, nullptr, latout, 4, 128, 64, 64, 64, 1, 1, 0, false, CF_RELU_IN);
    };

    encoder(Zin, 128, EZ_C1W, EZ_C1B, EZ_C2W, EZ_C2B, 3, 1, EZ_C3W, EZ_C3B,
            EZ_R1, EZ_R2, LAT);
    const float* xs[3] = {X1, X2, X3};
    for (int i = 0; i < 3; ++i)
        encoder(xs[i], 256, EX_C1W, EX_C1B, EX_C2W, EX_C2B, 4, 2, EX_C3W, EX_C3B,
                EX_R1, EX_R2, LAT + (size_t)(1 + i) * 1048576);

    // ---------------- Vector quantization ----------------
    for (int i = 0; i < 4; ++i) {
        const float* book = (i == 0) ? EMBZ : EMB;
        const float* es   = (i == 0) ? ESQZ : ESQ;
        const float* lat  = LAT + (size_t)i * 1048576;
        float* zq = ZQ + (size_t)i * 1048576;
        // scores = <z,e>: 1x1 conv with the codebook as a 512x64 weight matrix
        conv(lat, book, nullptr, nullptr, BUF1, 4, 64, 64, 512, 64, 1, 1, 0, false, 0);
        vq_argmin_kernel<<<64, 256, 0, stream>>>(BUF1, es, IDX, COUNTS + i * 512);
        vq_gather_kernel<<<4096, 256, 0, stream>>>(lat, book, IDX, zq, ACC + 0);
    }

    fusion_kernel<<<4096, 256, 0, stream>>>(ZQ, MU1, MU2, FUS, 1048576);

    // ---------------- Decoders ----------------
    auto decoder = [&](const float* lin,
                       const float* t1w, const float* t1b,
                       const float* r1, const float* r2,
                       const float* t2w, const float* t2b, int k2, int s2,
                       const float* t3w, const float* t3b,
                       const float* target, float* accslot) {
        conv(lin, t1w, t1b, nullptr, BUF3, 4, 64, 64, 128, 64, 3, 1, 1, true, 0);
        conv(BUF3, r1, nullptr, nullptr, BUF5, 4, 128, 64, 32, 64, 3, 1, 1, false, CF_RELU_IN);
        conv(BUF5, r2, nullptr, BUF3, BUF4, 4, 32, 64, 128, 64, 1, 1, 0, false, CF_RELU_IN);
        conv(BUF4, r1, nullptr, nullptr, BUF5, 4, 128, 64, 32, 64, 3, 1, 1, false, CF_RELU_IN);
        conv(BUF5, r2, nullptr, BUF4, BUF3, 4, 32, 64, 128, 64, 1, 1, 0, false, CF_RELU_IN);
        int H2 = (s2 == 2) ? 128 : 64;
        float* o2 = (s2 == 2) ? BUF0 : BUF2;
        conv(BUF3, t2w, t2b, nullptr, o2, 4, 128, 64, 256, H2, k2, s2, 1, true,
             CF_RELU_IN | CF_RELU_OUT);
        int H3 = H2 * 2;
        float* o3 = (s2 == 2) ? BUF2 : BUF5;
        conv(o2, t3w, t3b, nullptr, o3, 4, 256, H2, 3, H3, 4, 2, 1, true, 0);
        int n = 4 * 3 * H3 * H3;
        mse_kernel<<<(n + 255) / 256, 256, 0, stream>>>(target, o3, accslot, n);
    };

    decoder(FUS, D_T1W, D_T1B, D_R1, D_R2, D_T2W, D_T2B, 3, 1, D_T3W, D_T3B,
            ZTV, ACC + 1);                                   // L_t
    decoder(ZQ, DZ_T1W, DZ_T1B, DZ_R1, DZ_R2, DZ_T2W, DZ_T2B, 3, 1, DZ_T3W, DZ_T3B,
            Zin, ACC + 2);                                   // L_z
    decoder(ZQ + 1048576, DX_T1W, DX_T1B, DX_R1, DX_R2, DX_T2W, DX_T2B, 4, 2,
            DX_T3W, DX_T3B, X2, ACC + 3);                    // L_d term 1
    decoder(ZQ + 2097152, DX_T1W, DX_T1B, DX_R1, DX_R2, DX_T2W, DX_T2B, 4, 2,
            DX_T3W, DX_T3B, X3, ACC + 4);                    // L_d term 2

    finalize_kernel<<<1, 256, 0, stream>>>(ACC, COUNTS, OUT);
}
